// MultiHeadSelfAttention_81844896792826
// MI455X (gfx1250) — matrix operand order repaired and checked by reading, not compiled, not run on hardware
//
#include <hip/hip_runtime.h>

typedef unsigned short u16;
typedef __attribute__((ext_vector_type(16))) __bf16    v16bf;
typedef __attribute__((ext_vector_type(8)))  float     v8f;
typedef __attribute__((ext_vector_type(4)))  unsigned  v4u;

#define D_MODEL 1024
#define N_HEADS 16
#define HEAD_DIM 64
#define SEQ 2048
#define BATCH 2
#define M_TOTAL (BATCH * SEQ) /* 4096 */

__device__ __forceinline__ u16 f32_to_bf16(float f) {
  unsigned u = __float_as_uint(f);
  u += 0x7FFFu + ((u >> 16) & 1u);   // round-to-nearest-even
  return (u16)(u >> 16);
}

union FragBF {
  v16bf v;
  v4u   u[2];
};

// A-operand fragment (16x32 bf16, M x K), tile at `tile`, row stride ld (elems).
// ISA layout: lane L -> row L&15; elems 0..7 = K (L>=16?8:0)+0..7 ; elems 8..15 = +16.
__device__ __forceinline__ v16bf load_a_frag(const u16* tile, int ld) {
  const int lane = threadIdx.x & 31;
  const int r  = lane & 15;
  const int kh = (lane & 16) ? 8 : 0;
  const u16* p = tile + (size_t)r * ld + kh;
  FragBF f;
  f.u[0] = *(const v4u*)(p);
  f.u[1] = *(const v4u*)(p + 16);
  return f.v;
}

// B-operand fragment (32x16 bf16, K x N), stored transposed (N x K), row stride ld.
// ISA layout: lane L -> col L&15; elems 0..7 = K (L>=16?8:0)+0..7 ; elems 8..15 = +16.
__device__ __forceinline__ v16bf load_b_frag(const u16* tile, int ld) {
  const int lane = threadIdx.x & 31;
  const int c  = lane & 15;
  const int kh = (lane & 16) ? 8 : 0;
  const u16* p = tile + (size_t)c * ld + kh;
  FragBF f;
  f.u[0] = *(const v4u*)(p);
  f.u[1] = *(const v4u*)(p + 16);
  return f.v;
}

// Async global->LDS copy, 16 bytes per lane (GLOBAL_LOAD_ASYNC_TO_LDS_B128, GV mode).
// Tracked by ASYNCcnt; pair with s_wait_asynccnt + block barrier before LDS reads.
__device__ __forceinline__ void async_load_b128(const void* gptr, void* lptr) {
  unsigned lds_addr = (unsigned)(size_t)lptr;                 // low 32 bits = LDS offset
  unsigned long long gaddr = (unsigned long long)(size_t)gptr;
  asm volatile("global_load_async_to_lds_b128 %0, %1, off"
               :
               : "v"(lds_addr), "v"(gaddr)
               : "memory");
}

__device__ __forceinline__ void wait_async_all() {
  asm volatile("s_wait_asynccnt 0" ::: "memory");
}

__global__ __launch_bounds__(256) void cast_bf16_kernel(const float* __restrict__ in,
                                                        u16* __restrict__ out, int n) {
  int i = blockIdx.x * 256 + threadIdx.x;
  if (i < n) out[i] = f32_to_bf16(in[i]);
}

// in: (K x N) row-major fp32  ->  out: (N x K) row-major bf16
__global__ __launch_bounds__(256) void transpose_cast_kernel(const float* __restrict__ in,
                                                             u16* __restrict__ out,
                                                             int K, int N) {
  int i = blockIdx.x * 256 + threadIdx.x;
  if (i >= K * N) return;
  int nn = i / K;
  int kk = i - nn * K;
  out[i] = f32_to_bf16(in[(size_t)kk * N + nn]);
}

// C(M x N) = A(M x K, bf16 row-major) * Bt(N x K, bf16 row-major == B transposed)
// Block = 8 waves: block tile 128(M) x 64(N). Bt k-slab staged in LDS via async copy,
// shared by all waves; each wave computes a 16x64 strip (A fragment reused 4x).
template <int OUT_BF16>
__global__ __launch_bounds__(256) void gemm_kernel(const u16* __restrict__ A,
                                                   const u16* __restrict__ Bt,
                                                   void* __restrict__ Cout,
                                                   int M, int N, int K) {
  __shared__ __align__(16) u16 bbuf[64 * 32];   // [n-local][k-local]

  const int lane = threadIdx.x & 31;
  const int wave = threadIdx.x >> 5;
  const int nblock = blockIdx.x * 64;
  const int m0 = blockIdx.y * 128 + wave * 16;
  const u16* a0 = A + (size_t)m0 * K;

  const int tid  = threadIdx.x;
  const int brow = tid >> 2;          // 0..63: local n row
  const int bch  = (tid & 3) * 8;     // 0,8,16,24: k chunk (8 elems = 16B)
  const u16* gB = Bt + (size_t)(nblock + brow) * K + bch;
  u16* lB = &bbuf[brow * 32 + bch];

  v8f acc0 = {}, acc1 = {}, acc2 = {}, acc3 = {};
  for (int k0 = 0; k0 < K; k0 += 32) {
    __syncthreads();                       // previous iteration done with bbuf
    async_load_b128(gB + k0, lB);          // stage Bt[nblock..+63][k0..+31]
    if (k0 + 32 < K) __builtin_prefetch(a0 + k0 + 32, 0, 3);
    v16bf a = load_a_frag(a0 + k0, K);     // overlap with async copy
    wait_async_all();
    __syncthreads();
    acc0 = __builtin_amdgcn_wmma_f32_16x16x32_bf16(false, a, false,
              load_b_frag(bbuf + 0 * 16 * 32, 32), (short)0, acc0, false, false);
    acc1 = __builtin_amdgcn_wmma_f32_16x16x32_bf16(false, a, false,
              load_b_frag(bbuf + 1 * 16 * 32, 32), (short)0, acc1, false, false);
    acc2 = __builtin_amdgcn_wmma_f32_16x16x32_bf16(false, a, false,
              load_b_frag(bbuf + 2 * 16 * 32, 32), (short)0, acc2, false, false);
    acc3 = __builtin_amdgcn_wmma_f32_16x16x32_bf16(false, a, false,
              load_b_frag(bbuf + 3 * 16 * 32, 32), (short)0, acc3, false, false);
  }

  const int col = nblock + (lane & 15);
  const int r0  = m0 + ((lane & 16) ? 8 : 0);
  if (OUT_BF16) {
    u16* C = (u16*)Cout;
#pragma unroll
    for (int v = 0; v < 8; ++v) {
      u16* crow = C + (size_t)(r0 + v) * N + col;
      crow[0]  = f32_to_bf16(acc0[v]);
      crow[16] = f32_to_bf16(acc1[v]);
      crow[32] = f32_to_bf16(acc2[v]);
      crow[48] = f32_to_bf16(acc3[v]);
    }
  } else {
    float* C = (float*)Cout;
#pragma unroll
    for (int v = 0; v < 8; ++v) {
      float* crow = C + (size_t)(r0 + v) * N + col;
      crow[0]  = acc0[v];
      crow[16] = acc1[v];
      crow[32] = acc2[v];
      crow[48] = acc3[v];
    }
  }
}

// Flash attention over qkv (M_TOTAL x 3*D_MODEL bf16). Output attnb (M_TOTAL x D_MODEL bf16).
// grid = (SEQ/128, N_HEADS, BATCH); 8 waves/block, 16 query rows per wave.
__global__ __launch_bounds__(256) void attn_kernel(const u16* __restrict__ qkv,
                                                   u16* __restrict__ attnb) {
  __shared__ __align__(16) u16 kbuf[32 * 64];        // K tile: [key s][d]
  __shared__ __align__(16) u16 vt[64 * 32];          // V tile transposed: [d][key s]
  __shared__ __align__(16) u16 pbuf[8][16 * 32];     // per-wave P tile: [row][key s]

  const int lane = threadIdx.x & 31;
  const int wave = threadIdx.x >> 5;
  const int b = blockIdx.z;
  const int h = blockIdx.y;
  const int m0 = blockIdx.x * 128 + wave * 16;
  const int rowstride = 3 * D_MODEL;

  // Q fragments (loaded once): rows m0..m0+15, d 0..31 and 32..63
  const u16* qbase = qkv + (size_t)(b * SEQ + m0) * rowstride + h * HEAD_DIM;
  const v16bf qa0 = load_a_frag(qbase, rowstride);
  const v16bf qa1 = load_a_frag(qbase + 32, rowstride);

  v8f acc0 = {}, acc1 = {}, acc2 = {}, acc3 = {};
  float mrow[8], lrow[8];
#pragma unroll
  for (int v = 0; v < 8; ++v) { mrow[v] = -1e30f; lrow[v] = 0.0f; }

  u16* mypbuf = &pbuf[wave][0];
  const int tid = threadIdx.x;
  const int ls  = tid >> 3;        // 0..31: key row in tile
  const int lch = (tid & 7) * 8;   // 0..56: d offset (8 elems each)

  for (int j0 = 0; j0 < SEQ; j0 += 32) {
    __syncthreads();
    // K tile: async copy straight into LDS (row-major along d).
    const size_t srow = (size_t)(b * SEQ + j0 + ls) * rowstride + h * HEAD_DIM + lch;
    async_load_b128(qkv + srow + D_MODEL, &kbuf[ls * 64 + lch]);
    // V tile: load + transpose into LDS [d][s] for the PV B-operand.
    v4u vv = *(const v4u*)(qkv + srow + 2 * D_MODEL);
    u16 vs[8];
    __builtin_memcpy(vs, &vv, 16);
#pragma unroll
    for (int e = 0; e < 8; ++e) vt[(lch + e) * 32 + ls] = vs[e];
    wait_async_all();
    __syncthreads();

    // S = Q * K^T for 2 key tiles of 16, each over d = 2 x 32
    v8f s0 = {}, s1 = {};
    s0 = __builtin_amdgcn_wmma_f32_16x16x32_bf16(false, qa0, false,
            load_b_frag(kbuf, 64), (short)0, s0, false, false);
    s0 = __builtin_amdgcn_wmma_f32_16x16x32_bf16(false, qa1, false,
            load_b_frag(kbuf + 32, 64), (short)0, s0, false, false);
    s1 = __builtin_amdgcn_wmma_f32_16x16x32_bf16(false, qa0, false,
            load_b_frag(kbuf + 16 * 64, 64), (short)0, s1, false, false);
    s1 = __builtin_amdgcn_wmma_f32_16x16x32_bf16(false, qa1, false,
            load_b_frag(kbuf + 16 * 64 + 32, 64), (short)0, s1, false, false);

    // Online softmax update; P written to LDS in [row][s] layout for A-frag regather.
    const int ccol = lane & 15;
    const int rhalf = (lane & 16) ? 8 : 0;
#pragma unroll
    for (int v = 0; v < 8; ++v) {
      float x0 = s0[v] * 0.125f;   // 1/sqrt(64)
      float x1 = s1[v] * 0.125f;
      float tm = fmaxf(x0, x1);
      tm = fmaxf(tm, __shfl_xor(tm, 1));
      tm = fmaxf(tm, __shfl_xor(tm, 2));
      tm = fmaxf(tm, __shfl_xor(tm, 4));
      tm = fmaxf(tm, __shfl_xor(tm, 8));
      float mnew  = fmaxf(mrow[v], tm);
      float alpha = __expf(mrow[v] - mnew);
      float p0 = __expf(x0 - mnew);
      float p1 = __expf(x1 - mnew);
      float rs = p0 + p1;
      rs += __shfl_xor(rs, 1);
      rs += __shfl_xor(rs, 2);
      rs += __shfl_xor(rs, 4);
      rs += __shfl_xor(rs, 8);
      lrow[v] = lrow[v] * alpha + rs;
      mrow[v] = mnew;
      acc0[v] *= alpha; acc1[v] *= alpha; acc2[v] *= alpha; acc3[v] *= alpha;
      const int r = v + rhalf;
      mypbuf[r * 32 + ccol]      = f32_to_bf16(p0);
      mypbuf[r * 32 + 16 + ccol] = f32_to_bf16(p1);
    }
    asm volatile("s_wait_dscnt 0" ::: "memory");   // wave-local LDS RAW fence

    const v16bf pa = load_a_frag(mypbuf, 32);
    acc0 = __builtin_amdgcn_wmma_f32_16x16x32_bf16(false, pa, false,
              load_b_frag(vt + 0 * 16 * 32, 32), (short)0, acc0, false, false);
    acc1 = __builtin_amdgcn_wmma_f32_16x16x32_bf16(false, pa, false,
              load_b_frag(vt + 1 * 16 * 32, 32), (short)0, acc1, false, false);
    acc2 = __builtin_amdgcn_wmma_f32_16x16x32_bf16(false, pa, false,
              load_b_frag(vt + 2 * 16 * 32, 32), (short)0, acc2, false, false);
    acc3 = __builtin_amdgcn_wmma_f32_16x16x32_bf16(false, pa, false,
              load_b_frag(vt + 3 * 16 * 32, 32), (short)0, acc3, false, false);
  }

  // Normalize and store bf16 attention output (row-major M x D_MODEL).
  const int col   = lane & 15;
  const int rbase = (lane & 16) ? 8 : 0;
#pragma unroll
  for (int v = 0; v < 8; ++v) {
    const float inv = 1.0f / lrow[v];
    u16* orow = attnb + (size_t)(b * SEQ + m0 + rbase + v) * D_MODEL + h * HEAD_DIM + col;
    orow[0]  = f32_to_bf16(acc0[v] * inv);
    orow[16] = f32_to_bf16(acc1[v] * inv);
    orow[32] = f32_to_bf16(acc2[v] * inv);
    orow[48] = f32_to_bf16(acc3[v] * inv);
  }
}

extern "C" void kernel_launch(void* const* d_in, const int* in_sizes, int n_in,
                              void* d_out, int out_size, void* d_ws, size_t ws_size,
                              hipStream_t stream) {
  const float* x      = (const float*)d_in[0];
  const float* w_qkv  = (const float*)d_in[1];
  const float* w_proj = (const float*)d_in[2];
  float* out = (float*)d_out;

  // Workspace layout (bf16 elements): 48 MB total
  u16* xb     = (u16*)d_ws;                                 // 4096 x 1024
  u16* wqkvT  = xb     + (size_t)M_TOTAL * D_MODEL;         // 3072 x 1024 (transposed)
  u16* wprojT = wqkvT  + (size_t)3 * D_MODEL * D_MODEL;     // 1024 x 1024 (transposed)
  u16* qkvb   = wprojT + (size_t)D_MODEL * D_MODEL;         // 4096 x 3072
  u16* attnb  = qkvb   + (size_t)M_TOTAL * 3 * D_MODEL;     // 4096 x 1024

  cast_bf16_kernel<<<(M_TOTAL * D_MODEL) / 256, 256, 0, stream>>>(x, xb, M_TOTAL * D_MODEL);
  transpose_cast_kernel<<<(3 * D_MODEL * D_MODEL) / 256, 256, 0, stream>>>(
      w_qkv, wqkvT, D_MODEL, 3 * D_MODEL);
  transpose_cast_kernel<<<(D_MODEL * D_MODEL) / 256, 256, 0, stream>>>(
      w_proj, wprojT, D_MODEL, D_MODEL);

  // qkv = x @ w_qkv   (bf16 out)
  gemm_kernel<1><<<dim3(3 * D_MODEL / 64, M_TOTAL / 128), 256, 0, stream>>>(
      xb, wqkvT, qkvb, M_TOTAL, 3 * D_MODEL, D_MODEL);

  // flash attention
  attn_kernel<<<dim3(SEQ / 128, N_HEADS, BATCH), 256, 0, stream>>>(qkvb, attnb);

  // out = attn @ w_proj   (fp32 out)
  gemm_kernel<0><<<dim3(D_MODEL / 64, M_TOTAL / 128), 256, 0, stream>>>(
      attnb, wprojT, out, M_TOTAL, D_MODEL, D_MODEL);
}
